// IrrepLinear_83451214561663
// MI455X (gfx1250) — matrix operand order repaired and checked by reading, not compiled, not run on hardware
//
#include <hip/hip_runtime.h>

// ---------------------------------------------------------------------------
// IrrepLinear on gfx1250 (MI455X): 5 grouped GEMMs via V_WMMA_F32_16X16X32_BF16
// with a 2-term bf16 split (hi+lo, 3 products) for near-fp32 accuracy.
// Round 6: merge all LDS regions into one array so every fragment load is
// (one base VGPR + 16-bit ds immediate). This removes the ~26 hoisted per-tile
// address registers that were spilling into the WMMA loop in rounds 4-5.
// ---------------------------------------------------------------------------

typedef __attribute__((ext_vector_type(16))) __bf16        v16bf;
typedef __attribute__((ext_vector_type(8)))  float         v8f;
typedef __attribute__((ext_vector_type(8)))  unsigned int  v8u;

#define IN_F   512
#define OUT_F  512
#define MDIM   25          // (LMAX+1)^2
#define BATCH  4096
#define NB     16          // batches per block -> 16*(2l+1) rows per l (all %16==0)
#define NO     128         // output features per block
#define KC     32          // K chunk (= one bf16-WMMA K)
#define KP     36          // padded row stride in halfwords (bank-conflict-free)
#define ROWS   (NB * MDIM) // 400 rows -> 25 tiles of 16, each uniform l
#define NKC    (IN_F / KC) // 16 chunks
#define NT     512         // threads per block (16 waves)
#define XTOT   3200        // NB*200 float4s of x per chunk
#define XPIPE  5           // register-pipelined float4s per thread (unguarded)

// one merged LDS allocation (halfword offsets), deterministic layout:
#define LSAH   0                         // A hi : 14400 hw (28.8 KB)
#define LSAL   (ROWS * KP)               // A lo : +28800 B (fits ds imm)
#define LSBH   (2 * ROWS * KP)           // B hi : +57600 B (fits ds imm)
#define LSBL   (2 * ROWS * KP + 5 * NO * KP) // B lo : +103680 B (needs 1 add)
#define LSTOT  (2 * ROWS * KP + 2 * 5 * NO * KP) // 74880 hw = 146.25 KB

// l(m): thresholds 1,4,9,16 (also valid for tile index since tile_start(l)=l^2)
__device__ __forceinline__ int l_of(int m) {
  return (m >= 16) ? 4 : (m >= 9) ? 3 : (m >= 4) ? 2 : (m >= 1) ? 1 : 0;
}

// f32 -> bf16 hi (RNE) + bf16 lo (RNE of exact residual)
__device__ __forceinline__ void split2(float f, unsigned short* hi, unsigned short* lo) {
  unsigned u  = __builtin_bit_cast(unsigned, f);
  unsigned rh = u + 0x7FFFu + ((u >> 16) & 1u);
  unsigned short h16 = (unsigned short)(rh >> 16);
  float fh = __builtin_bit_cast(float, (unsigned)h16 << 16);
  float fl = f - fh;                       // exact residual
  unsigned ul = __builtin_bit_cast(unsigned, fl);
  unsigned rl = ul + 0x7FFFu + ((ul >> 16) & 1u);
  *hi = h16;
  *lo = (unsigned short)(rl >> 16);
}

// A/B fragment gather at halfword offset `eoff` (base VGPR + folded constants).
// A layout: lane half h holds K in [8h,8h+8) U [8h+16,8h+24)  -> +{0,4,16,20}
__device__ __forceinline__ v16bf fragA(const unsigned short* ls, int eoff) {
  v8u u; uint2 q;
  q = *(const uint2*)(ls + eoff + 0);  u[0] = q.x; u[1] = q.y;   // K 8h+0..3
  q = *(const uint2*)(ls + eoff + 4);  u[2] = q.x; u[3] = q.y;   // K 8h+4..7
  q = *(const uint2*)(ls + eoff + 16); u[4] = q.x; u[5] = q.y;   // K 8h+16..19
  q = *(const uint2*)(ls + eoff + 20); u[6] = q.x; u[7] = q.y;   // K 8h+20..23
  return __builtin_bit_cast(v16bf, u);
}
// B layout: lane half h holds K in [8h,8h+8) U [8h+16,8h+24); N = lane%16
__device__ __forceinline__ v16bf fragB(const unsigned short* ls, int eoff) {
  v8u u; uint2 q;
  q = *(const uint2*)(ls + eoff + 0);  u[0] = q.x; u[1] = q.y;
  q = *(const uint2*)(ls + eoff + 4);  u[2] = q.x; u[3] = q.y;
  q = *(const uint2*)(ls + eoff + 16); u[4] = q.x; u[5] = q.y;
  q = *(const uint2*)(ls + eoff + 20); u[6] = q.x; u[7] = q.y;
  return __builtin_bit_cast(v16bf, u);
}

__global__ __launch_bounds__(NT)
void irrep_linear_wmma(const float* __restrict__ x,     // [B, IN_F*25]  (i-major, m-minor)
                       const float* __restrict__ wgt,   // [5, OUT_F, IN_F]
                       const float* __restrict__ bias,  // [OUT_F]
                       float* __restrict__ out) {       // [B, OUT_F*25] (o-major, m-minor)
  __shared__ unsigned short lsMem[LSTOT];

  const int tid  = threadIdx.x;
  const int wv   = __builtin_amdgcn_readfirstlane((int)(tid >> 5)); // scalar wave id
  const int lane = tid & 31;
  const int s    = lane & 15;   // row (A) / col (B,C) within tile
  const int h    = lane >> 4;   // lane half
  const int co   = (wv & 7) * 16;       // this wave's 16-col tile within NO=128
  const int half = wv >> 3;             // row-tile half: 0 -> tiles 0..12, 1 -> 13..24
  const int t0   = half ? 13 : 0;
  const int nt   = half ? 12 : 13;
  const int bg0  = blockIdx.x * NB;
  const int ob0  = blockIdx.y * NO;

  // single persistent A-frag base; per-tile deltas are compile-time immediates
  const int abase = (16 * t0 + s) * KP + 8 * h;
  // per-lane part of the B-frag base (array region + l*NO*KP added per l-run)
  const int bbase = (co + s) * KP + 8 * h;

  v8f acc[13];
#pragma unroll
  for (int i = 0; i < 13; ++i)
#pragma unroll
    for (int e = 0; e < 8; ++e) acc[i][e] = 0.0f;

  // ---- prologue: issue chunk-0 pipelined x loads (fly across first barrier) ----
  float4 xa[XPIPE];
#pragma unroll
  for (int i = 0; i < XPIPE; ++i) {        // i*NT + tid < 3200 always: no guards
    int idx = tid + i * NT;
    int b = idx / 200, q = idx - b * 200;
    xa[i] = *(const float4*)(x + (bg0 + b) * (IN_F * MDIM) + q * 4);
  }

  for (int kc = 0; kc < NKC; ++kc) {
    const int k0 = kc * KC;
    __syncthreads();   // previous chunk's frag reads done before overwriting LDS

    // ---- synchronous tail of this chunk's x: issue loads first, convert last ----
    float4 x5, x6;
    {
      int idx = tid + XPIPE * NT;                     // 2560..3071 < 3200: valid
      int b = idx / 200, q = idx - b * 200;
      x5 = *(const float4*)(x + (bg0 + b) * (IN_F * MDIM) + q * 4 + k0 * MDIM);
    }
    const bool has6 = tid < XTOT - 6 * NT;            // tid < 128
    if (has6) {
      int idx = tid + 6 * NT;
      int b = idx / 200, q = idx - b * 200;
      x6 = *(const float4*)(x + (bg0 + b) * (IN_F * MDIM) + q * 4 + k0 * MDIM);
    }

    // ---- store x registers as bf16 hi/lo into the l-regrouped LDS rows ----
#pragma unroll
    for (int i = 0; i < XPIPE + 2; ++i) {
      if (i == 6 && !has6) break;          // i<6 always valid
      int idx = idx = tid + i * NT;
      int b = idx / 200, q = idx - b * 200;
      const float4 v = (i < XPIPE) ? xa[i] : (i == XPIPE ? x5 : x6);
      float vv[4] = {v.x, v.y, v.z, v.w};
#pragma unroll
      for (int c = 0; c < 4; ++c) {
        int e  = q * 4 + c;
        int kk = e / 25;            // const divisor -> mul/shift
        int m  = e - kk * 25;
        int l  = l_of(m);
        int rl = 16 * l * l + b * (2 * l + 1) + (m - l * l); // regrouped by l
        unsigned short hi, lo;
        split2(vv[c], &hi, &lo);
        lsMem[LSAH + rl * KP + kk] = hi;   // lo shares address: +28800B immediate
        lsMem[LSAL + rl * KP + kk] = lo;
      }
    }

    // ---- stage weights for this chunk (L2-resident; 10 b128 loads issue together)
#pragma unroll
    for (int it = 0; it < 10; ++it) {            // 5*NO*8 / NT == 10 exactly
      int idx = tid + it * NT;
      int l = idx >> 10; int r = idx & 1023;     // NO*8 = 1024
      int o = r >> 3;    int q = r & 7;
      const float4 v = *(const float4*)(wgt + (((l << 9) + ob0 + o) << 9) + k0 + q * 4);
      float vv[4] = {v.x, v.y, v.z, v.w};
      int lb = (l * NO + o) * KP + q * 4;
#pragma unroll
      for (int c = 0; c < 4; ++c) {
        unsigned short hi, lo;
        split2(vv[c], &hi, &lo);
        lsMem[LSBH + lb + c] = hi;
        lsMem[LSBL + lb + c] = lo;
      }
    }

    // ---- issue next chunk's pipelined x loads; they overlap the WMMA section ----
    if (kc + 1 < NKC) {
#pragma unroll
      for (int i = 0; i < XPIPE; ++i) {
        int idx = tid + i * NT;
        int b = idx / 200, q = idx - b * 200;
        xa[i] = *(const float4*)(x + (bg0 + b) * (IN_F * MDIM) + q * 4
                                   + (kc + 1) * (KC * MDIM));
      }
    }

    __syncthreads();

    // ---- warm next chunk's weight lines in cache (no LOADcnt, no registers) ----
    if (kc + 1 < NKC) {
#pragma unroll
      for (int it = 0; it < 2; ++it) {           // 5*NO = 640 rows, 1 line each
        int idx = tid + it * NT;
        if (idx < 5 * NO) {
          int l = idx >> 7; int o = idx & (NO - 1);
          __builtin_prefetch((const void*)(wgt + (((l << 9) + ob0 + o) << 9)
                                               + k0 + KC), 0, 3);
        }
      }
    }

    // ---- compute: 13 (or 12) row tiles x 1 col tile, 3 WMMAs each ----
    v16bf bh{}, bl{};
    int lcur = -1;
#pragma unroll
    for (int tt = 0; tt < 13; ++tt) {
      if (tt < nt) {                       // scalar condition: EXEC stays all-ones
        int t = t0 + tt;
        int l = l_of(t);                   // tile -> l (tile_start(l) = l*l)
        if (l != lcur) {                   // scalar: reload B frags on l change
          int bo = l * (NO * KP) + bbase;  // transient per l-run
          bh = fragB(lsMem + LSBH, bo);
          bl = fragB(lsMem + LSBL, bo);
          lcur = l;
        }
        // one base VGPR + compile-time immediates for all tiles, hi and lo:
        v16bf ah = fragA(lsMem + LSAH, abase + tt * (16 * KP));
        v16bf al = fragA(lsMem + LSAL, abase + tt * (16 * KP));
        acc[tt] = __builtin_amdgcn_wmma_f32_16x16x32_bf16(false, ah, false, bh,
                                                          (short)0, acc[tt], false, false);
        acc[tt] = __builtin_amdgcn_wmma_f32_16x16x32_bf16(false, ah, false, bl,
                                                          (short)0, acc[tt], false, false);
        acc[tt] = __builtin_amdgcn_wmma_f32_16x16x32_bf16(false, al, false, bh,
                                                          (short)0, acc[tt], false, false);
      }
    }
  }

  // ---- epilogue: C/D layout lane L: N = L%16, M = v + 8*(L/16) ----
  const int oG = ob0 + co + s;
  const float bs = bias[oG];
#pragma unroll
  for (int tt = 0; tt < 13; ++tt) {
    if (tt < nt) {
      int t    = t0 + tt;
      int l    = l_of(t);
      int step = 2 * l + 1;
      int base = 16 * l * l;
#pragma unroll
      for (int v = 0; v < 8; ++v) {
        int M   = v + 8 * h;
        int rg  = 16 * t + M - base;       // row index within l group
        int bl_ = rg / step;               // batch-local
        int m   = l * l + (rg - bl_ * step);
        float val = acc[tt][v];
        if (t == 0) val += bs;             // bias only on m==0 (the whole l=0 tile)
        out[(size_t)(bg0 + bl_) * (OUT_F * MDIM) + (size_t)oG * MDIM + m] = val;
      }
    }
  }
}

extern "C" void kernel_launch(void* const* d_in, const int* in_sizes, int n_in,
                              void* d_out, int out_size, void* d_ws, size_t ws_size,
                              hipStream_t stream) {
  (void)in_sizes; (void)n_in; (void)d_ws; (void)ws_size; (void)out_size;
  const float* x  = (const float*)d_in[0];
  const float* w  = (const float*)d_in[1];
  const float* b  = (const float*)d_in[2];
  float* out      = (float*)d_out;
  dim3 grid(BATCH / NB, OUT_F / NO);   // (256, 4)
  irrep_linear_wmma<<<grid, dim3(NT), 0, stream>>>(x, w, b, out);
}
